// CGLayers_72138270703993
// MI455X (gfx1250) — matrix operand order repaired and checked by reading, not compiled, not run on hardware
//
#include <hip/hip_runtime.h>
#include <hip/hip_bf16.h>
#include <stdint.h>

// ---------------------------------------------------------------------------
// Problem constants (fixed by the reference): B=2, N=192, C=16, MAX_L=2.
// Feature layout per node: 9 "m" rows (l=0: row 0, l=1: rows 1..3, l=2: rows 4..8)
// x 16 channels = 144 floats per node.
// ---------------------------------------------------------------------------
#define BB     2
#define NN     192
#define CC     16
#define MM     9
#define FDIM   144          // MM*CC
#define NODES  (BB * NN)    // 384

typedef __attribute__((ext_vector_type(16))) _Float16 v16h;
typedef __attribute__((ext_vector_type(8)))  _Float16 v8h;
typedef __attribute__((ext_vector_type(8)))  float    v8f;
typedef __attribute__((ext_vector_type(4)))  unsigned u32x4;
typedef __attribute__((ext_vector_type(8)))  int      i32x8;
typedef __attribute__((ext_vector_type(4)))  int      i32x4;

// Use the Tensor Data Mover for the CG-table LDS staging when the ROCm 7.2
// (clang-22) 5-arg builtin is available; otherwise fall back to a manual copy.
#if defined(__has_builtin)
#  if __has_builtin(__builtin_amdgcn_tensor_load_to_lds) && (__clang_major__ < 23)
#    define USE_TDM 1
#  endif
#endif
#ifndef USE_TDM
#  define USE_TDM 0
#endif

// Swizzled-weight region layout (per layer, in halfs):
//   nl0 @ 0 (K=768), nl1 @ 12288 (K=1536), nl2 @ 36864 (K=1536),
//   rl0 @ 61440 (Kp=64), rl1 @ 62464 (Kp=96), rl2 @ 64000 (Kp=96); total 65536.
#define WT_LAYER 65536

// ---------------------------------------------------------------------------
// Clebsch-Gordan table init (Racah formula), written once into workspace.
// Index: ((((l1*3+l2)*3+l)*5 + i1)*5 + i2)*5 + i3, i = m + l.  3375 floats.
// ---------------------------------------------------------------------------
__device__ __forceinline__ double dfact(int n) {
    double f = 1.0;
    for (int i = 2; i <= n; ++i) f *= (double)i;
    return f;
}

__global__ void k_init_cg(float* cg) {
    int tid = threadIdx.x;
    for (int i = tid; i < 3375; i += 256) cg[i] = 0.0f;
    __syncthreads();
    for (int idx = tid; idx < 675; idx += 256) {
        int i2 = idx % 5;
        int i1 = (idx / 5) % 5;
        int l  = (idx / 25) % 3;
        int l2 = (idx / 75) % 3;
        int l1 = idx / 225;
        int lo = (l1 > l2) ? (l1 - l2) : (l2 - l1);
        int hi = (l1 + l2 < 2) ? (l1 + l2) : 2;
        if (l < lo || l > hi) continue;
        if (i1 > 2 * l1 || i2 > 2 * l2) continue;
        int m1 = i1 - l1, m2 = i2 - l2, m = m1 + m2;
        if (m < -l || m > l) continue;
        double pref = sqrt((2.0 * l + 1.0) * dfact(l1 + l2 - l) * dfact(l1 - l2 + l)
                           * dfact(-l1 + l2 + l) / dfact(l1 + l2 + l + 1));
        pref *= sqrt(dfact(l1 + m1) * dfact(l1 - m1) * dfact(l2 + m2)
                     * dfact(l2 - m2) * dfact(l + m) * dfact(l - m));
        double s = 0.0;
        for (int k = 0; k <= l1 + l2 - l; ++k) {
            int d0 = k, d1 = l1 + l2 - l - k, d2 = l1 - m1 - k, d3 = l2 + m2 - k;
            int d4 = l - l2 + m1 + k, d5 = l - l1 - m2 + k;
            if (d0 < 0 || d1 < 0 || d2 < 0 || d3 < 0 || d4 < 0 || d5 < 0) continue;
            s += ((k & 1) ? -1.0 : 1.0)
                 / (dfact(d0) * dfact(d1) * dfact(d2) * dfact(d3) * dfact(d4) * dfact(d5));
        }
        cg[((((l1 * 3 + l2) * 3 + l) * 5 + i1) * 5 + i2) * 5 + (m + l)] = (float)(pref * s);
    }
}

// ---------------------------------------------------------------------------
// Weight pre-swizzle: w[tau][16] f32  ->  WT[c][k] f16, K padded to `taup`.
// Makes every WMMA B-fragment 32 contiguous bytes per lane.
// ---------------------------------------------------------------------------
__global__ void k_prep_w(const float* __restrict__ src, _Float16* __restrict__ dst,
                         int tau, int taup) {
    int i = blockIdx.x * blockDim.x + threadIdx.x;
    if (i >= CC * taup) return;
    int c = i / taup, k = i % taup;
    dst[i] = (k < tau) ? (_Float16)src[k * CC + c] : (_Float16)0.0f;
}

// conn16[b,i,j] = (norms < 0.5) ? 1 : 0 in f16 (exact).
__global__ void k_prep_conn(const float* __restrict__ norms, _Float16* __restrict__ conn16) {
    int i = blockIdx.x * blockDim.x + threadIdx.x;
    if (i >= BB * NN * NN) return;
    conn16[i] = (_Float16)((norms[i] < 0.5f) ? 1.0f : 0.0f);
}

// ---------------------------------------------------------------------------
// Pack vert0/vert1/vert2 into P[node][f] (f32) and PT16[b][f][j] (f16, j-major)
// so the message-pass B fragment is a single 32-byte contiguous load per lane.
// ---------------------------------------------------------------------------
__global__ void k_pack(const float* v0, const float* v1, const float* v2,
                       float* P, _Float16* PT16) {
    int i = blockIdx.x * blockDim.x + threadIdx.x;
    if (i >= NODES * FDIM) return;
    int node = i / FDIM;
    int f = i % FDIM;
    int m = f / CC, c = f % CC;
    float v;
    if (m == 0)      v = v0[node * CC + c];
    else if (m < 4)  v = v1[node * 3 * CC + (m - 1) * CC + c];
    else             v = v2[node * 5 * CC + (m - 4) * CC + c];
    P[i] = v;
    int b = node / NN, j = node % NN;
    PT16[(b * FDIM + f) * NN + j] = (_Float16)v;
}

// ---------------------------------------------------------------------------
// Ysum[node][k] = sum_j Y_k(rel_pos[b,i,j]).  The j-sum of the relational CG
// pass commutes (only sph depends on j) -> exact O(N^2) -> O(N) factorization.
// ---------------------------------------------------------------------------
__global__ void k_ysum(const float* rel, float* ysum) {
    int row = blockIdx.x;
    int tid = threadIdx.x;      // 64
    float acc[9];
#pragma unroll
    for (int k = 0; k < 9; ++k) acc[k] = 0.0f;
    for (int j = tid; j < NN; j += 64) {
        const float* rp = &rel[(row * NN + j) * 3];
        float x = rp[0], y = rp[1], z = rp[2];
        float r = sqrtf(x * x + y * y + z * z + 1e-6f);
        float nx = x / r, ny = y / r, nz = z / r;
        acc[0] += 0.28209479177387814f;
        acc[1] += 0.4886025119029199f * ny;
        acc[2] += 0.4886025119029199f * nz;
        acc[3] += 0.4886025119029199f * nx;
        acc[4] += 1.0925484305920792f * nx * ny;
        acc[5] += 1.0925484305920792f * ny * nz;
        acc[6] += 0.31539156525252005f * (3.0f * nz * nz - 1.0f);
        acc[7] += 1.0925484305920792f * nx * nz;
        acc[8] += 0.5462742152960396f * (nx * nx - ny * ny);
    }
    __shared__ float sh[64 * 9];
#pragma unroll
    for (int k = 0; k < 9; ++k) sh[tid * 9 + k] = acc[k];
    __syncthreads();
    if (tid < 9) {
        float s = 0.0f;
        for (int t = 0; t < 64; ++t) s += sh[t * 9 + tid];
        ysum[row * 9 + tid] = s;
    }
}

// ---------------------------------------------------------------------------
// Message pass via WMMA: mp = conn @ parts.  A from conn16 (two aligned b128
// loads per lane), B from PT16 (one 32-byte vector load), f32 accumulate.
// ---------------------------------------------------------------------------
__global__ __launch_bounds__(32) void k_msgpass(const _Float16* __restrict__ conn16,
                                                const _Float16* __restrict__ PT16,
                                                float* __restrict__ mp) {
    int bidx = blockIdx.x;                // B * 12 * 9 tiles
    int ct = bidx % 9;
    int it = (bidx / 9) % 12;
    int b  = bidx / (9 * 12);
    int lane = threadIdx.x;
    int hl = lane >> 4;
    int rc = lane & 15;

    const _Float16* cr = conn16 + (size_t)(b * NN + it * 16 + rc) * NN;
    const _Float16* pr = PT16 + (size_t)(b * FDIM + ct * CC + rc) * NN;

    v8f acc = {};
    for (int kb = 0; kb < NN; kb += 32) {
        v8h r0 = *(const v8h*)(cr + kb + hl * 16);        // kk = hl*16 + 0..7
        v8h r1 = *(const v8h*)(cr + kb + hl * 16 + 8);    // kk = hl*16 + 8..15
        v16h a;
#pragma unroll
        for (int e = 0; e < 8; ++e) { a[e] = r0[e]; a[8 + e] = r1[e]; }
        v16h bb = *(const v16h*)(pr + kb + hl * 16);      // K = kb + hl*16 + 0..15
        acc = __builtin_amdgcn_wmma_f32_16x16x32_f16(false, a, false, bb, (short)0, acc,
                                                     false, false);
    }
#pragma unroll
    for (int r = 0; r < 8; ++r) {
        int row = hl * 8 + r;
        mp[(b * NN + it * 16 + row) * FDIM + ct * CC + rc] = acc[r];
    }
}

// ---------------------------------------------------------------------------
// Per-node kernel: 3 waves, one per output l.
// CG table staged to LDS by the Tensor Data Mover (s_wait_tensorcnt), then
//  phase 1: nl CG self-product built directly in the WMMA A layout, mixed
//           with swizzled w_nl (K = 768/1536/1536),
//  phase 2: relational CG product vs Ysum (j-sum prefactorized), mixed with
//           swizzled w_rel (Kp = 64/96/96).
// ---------------------------------------------------------------------------
__global__ __launch_bounds__(96) void k_node(
    const float* __restrict__ mp, const float* __restrict__ ysum,
    const float* __restrict__ cgg, const _Float16* __restrict__ WT,
    float* __restrict__ outp) {
    __shared__ float cg[3375];
    __shared__ float xl[FDIM];
    __shared__ float md[FDIM];

    int node = blockIdx.x;
    int tid = threadIdx.x;
    int wave = tid >> 5;
    int lane = tid & 31;
    int hl = lane >> 4, rc = lane & 15;

#if USE_TDM
    if (tid == 0) {
        unsigned lds_lo = (unsigned)(uintptr_t)(void*)cg;          // LDS byte offset
        unsigned long long ga = (unsigned long long)(uintptr_t)cgg;
        u32x4 g0 = { 1u,                                           // count = 1
                     lds_lo,                                       // lds_addr
                     (unsigned)ga,                                 // global_addr lo
                     (unsigned)(ga >> 32) | 0x80000000u };         // addr hi | type=2
        i32x8 g1 = { (int)(2u << 16),        // data_size = 4B
                     (int)(3375u << 16),     // tensor_dim0[15:0] << 16
                     (int)(1u << 16),        // tensor_dim0 hi | tensor_dim1 = 1
                     (int)(3375u << 16),     // tensor_dim1 hi | tile_dim0 = 3375
                     1,                      // tile_dim1 = 1
                     3375,                   // tensor_dim0_stride lo32
                     (int)(3375u << 16),     // stride hi | tensor_dim1_stride lo16
                     0 };
        i32x4 gz = { 0, 0, 0, 0 };
        __builtin_amdgcn_tensor_load_to_lds(g0, g1, gz, gz, 0);
    }
    if (wave == 0) __builtin_amdgcn_s_wait_tensorcnt(0);
#else
    for (int i = tid; i < 3375; i += 96) cg[i] = cgg[i];
#endif
    for (int i = tid; i < FDIM; i += 96) xl[i] = mp[node * FDIM + i];
    __syncthreads();

    const int rowoff[3] = {0, 1, 4};
    const int pl1[3][6] = {{0, 1, 2, 0, 0, 0}, {0, 1, 1, 1, 2, 2}, {0, 1, 1, 2, 2, 2}};
    const int pl2[3][6] = {{0, 1, 2, 0, 0, 0}, {1, 0, 1, 2, 1, 2}, {2, 1, 2, 0, 1, 2}};
    const int tau_nl[3]  = {768, 1536, 1536};
    const int off_nl[3]  = {0, 12288, 36864};
    const int tau_rl[3]  = {48, 96, 96};
    const int taup_rl[3] = {64, 96, 96};
    const int off_rl[3]  = {61440, 62464, 64000};

    int l = wave;                // one wave per output l (uniform per wave)
    int mrows = 2 * l + 1;

    // ---- phase 1: nonlinear CG self-product + channel mix ----
    {
        const _Float16* w = WT + off_nl[l] + rc * tau_nl[l];
        v8f acc = {};
        for (int kb = 0; kb < tau_nl[l]; kb += 32) {
            __builtin_prefetch(w + kb + 32 + hl * 16, 0, 1);
            v16h a;
#pragma unroll
            for (int e = 0; e < 16; ++e) {
                int vg = e >> 1, h = e & 1;
                int kk = hl * 16 + e;
                int t = kb + kk;          // t = pair*256 + c1*16 + c2
                float v = 0.0f;
                int m = rc;
                if (m < mrows) {
                    int pr2 = t >> 8, c1 = (t >> 4) & 15, c2 = t & 15;
                    int l1 = pl1[l][pr2], l2 = pl2[l][pr2];
                    for (int p = 0; p <= 2 * l1; ++p) {
                        int q = m - l - p + l1 + l2;   // m-selection: one q per p
                        if (q >= 0 && q <= 2 * l2) {
                            float cgv = cg[((((l1 * 3 + l2) * 3 + l) * 5 + p) * 5 + q) * 5 + m];
                            v += cgv * xl[(rowoff[l1] + p) * CC + c1]
                                     * xl[(rowoff[l2] + q) * CC + c2];
                        }
                    }
                }
                a[e] = (_Float16)v;
            }
            v16h bb = *(const v16h*)(w + kb + hl * 16);
            acc = __builtin_amdgcn_wmma_f32_16x16x32_f16(false, a, false, bb, (short)0, acc,
                                                         false, false);
        }
#pragma unroll
        for (int r = 0; r < 8; ++r) {
            int row = hl * 8 + r;
            if (row < mrows) md[(rowoff[l] + row) * CC + rc] = acc[r];
        }
    }
    __syncthreads();

    // ---- phase 2: relational CG product vs Ysum + channel mix ----
    {
        const _Float16* w = WT + off_rl[l] + rc * taup_rl[l];
        const float* ys = &ysum[node * 9];
        int tau = tau_rl[l];
        v8f acc = {};
        for (int kb = 0; kb < taup_rl[l]; kb += 32) {
            v16h a;
#pragma unroll
            for (int e = 0; e < 16; ++e) {
                int vg = e >> 1, h = e & 1;
                int kk = hl * 16 + e;
                int t = kb + kk;          // t = pair*16 + c1
                float v = 0.0f;
                int m = rc;
                if (m < mrows && t < tau) {
                    int pr2 = t >> 4, c1 = t & 15;
                    int l1 = pl1[l][pr2], l2 = pl2[l][pr2];
                    for (int p = 0; p <= 2 * l1; ++p) {
                        int q = m - l - p + l1 + l2;
                        if (q >= 0 && q <= 2 * l2) {
                            float cgv = cg[((((l1 * 3 + l2) * 3 + l) * 5 + p) * 5 + q) * 5 + m];
                            v += cgv * md[(rowoff[l1] + p) * CC + c1] * ys[rowoff[l2] + q];
                        }
                    }
                }
                a[e] = (_Float16)v;
            }
            v16h bb = *(const v16h*)(w + kb + hl * 16);   // zero-padded beyond tau
            acc = __builtin_amdgcn_wmma_f32_16x16x32_f16(false, a, false, bb, (short)0, acc,
                                                         false, false);
        }
#pragma unroll
        for (int r = 0; r < 8; ++r) {
            int row = hl * 8 + r;
            if (row < mrows) outp[node * FDIM + (rowoff[l] + row) * CC + rc] = acc[r];
        }
    }
}

// ---------------------------------------------------------------------------
// Per-l global sum of squares -> scale[l] = rsqrt(sum/NUM_CH).
// ---------------------------------------------------------------------------
__global__ void k_norm(const float* ps, float* scale) {
    int l = blockIdx.x;
    int tid = threadIdx.x;
    const int rowoff[3] = {0, 1, 4};
    int mr = 2 * l + 1;
    int total = NODES * mr * CC;
    float s = 0.0f;
    for (int i = tid; i < total; i += 256) {
        int node = i / (mr * CC);
        int rem = i % (mr * CC);
        int m = rem / CC, c = rem % CC;
        float v = ps[node * FDIM + (rowoff[l] + m) * CC + c];
        s += v * v;
    }
    __shared__ float sh[256];
    sh[tid] = s;
    __syncthreads();
    for (int o = 128; o > 0; o >>= 1) {
        if (tid < o) sh[tid] += sh[tid + o];
        __syncthreads();
    }
    if (tid == 0) scale[l] = rsqrtf(sh[0] / (float)CC);
}

// Apply normalization; also refresh the f16 transposed parts for the next
// layer's message pass.
__global__ void k_scale(const float* ps, const float* scale, float* P, _Float16* PT16) {
    int i = blockIdx.x * blockDim.x + threadIdx.x;
    if (i >= NODES * FDIM) return;
    int f = i % FDIM;
    int m = f / CC;
    int l = (m == 0) ? 0 : ((m < 4) ? 1 : 2);
    float v = ps[i] * scale[l];
    P[i] = v;
    int node = i / FDIM;
    int b = node / NN, j = node % NN;
    PT16[(b * FDIM + f) * NN + j] = (_Float16)v;
}

// ---------------------------------------------------------------------------
// SO(3)-invariant scalars: s0 (16) + per-l channel-pair Gram (3*256) = 784/layer.
// ---------------------------------------------------------------------------
__global__ void k_scalars(const float* P, float* out, int layer) {
    int node = blockIdx.x;
    int tid = threadIdx.x;                // 256 = (c1,c2)
    const int rowoff[3] = {0, 1, 4};
    float* o = out + node * 1568 + layer * 784;
    if (tid < 16) o[tid] = P[node * FDIM + tid];
    int c1 = tid >> 4, c2 = tid & 15;
#pragma unroll
    for (int l = 0; l < 3; ++l) {
        float s = 0.0f;
        for (int m = 0; m < 2 * l + 1; ++m) {
            const float* row = &P[node * FDIM + (rowoff[l] + m) * CC];
            s += row[c1] * row[c2];
        }
        o[16 + l * 256 + tid] = s;
    }
}

// ---------------------------------------------------------------------------
// Host launcher
// ---------------------------------------------------------------------------
extern "C" void kernel_launch(void* const* d_in, const int* in_sizes, int n_in,
                              void* d_out, int out_size, void* d_ws, size_t ws_size,
                              hipStream_t stream) {
    (void)in_sizes; (void)n_in; (void)out_size; (void)ws_size;
    const float* vert0  = (const float*)d_in[0];
    const float* vert1  = (const float*)d_in[1];
    const float* vert2  = (const float*)d_in[2];
    const float* rel    = (const float*)d_in[3];
    const float* norms  = (const float*)d_in[4];
    float* out = (float*)d_out;

    float* ws    = (float*)d_ws;
    float* cg    = ws;                          // 3375 f32
    float* ysum  = ws + 4096;                   // 3456 f32
    float* P     = ws + 8192;                   // 55296 f32
    float* mp    = ws + 65536;                  // 55296 f32
    float* ps    = ws + 131072;                 // 55296 f32
    float* scale = ws + 190464;                 // 3 f32
    _Float16* conn16 = (_Float16*)(ws + 190976); // 73728 f16
    _Float16* WT16   = (_Float16*)(ws + 229376); // 2 * 65536 f16 (swizzled weights)
    _Float16* PT16   = (_Float16*)(ws + 294912); // 55296 f16 (transposed parts)

    k_init_cg<<<1, 256, 0, stream>>>(cg);
    k_prep_conn<<<(BB * NN * NN + 255) / 256, 256, 0, stream>>>(norms, conn16);

    const int tau_nl[3] = {768, 1536, 1536};
    const int off_nl[3] = {0, 12288, 36864};
    const int tau_rl[3] = {48, 96, 96};
    const int taup_rl[3] = {64, 96, 96};
    const int off_rl[3] = {61440, 62464, 64000};
    for (int L = 0; L < 2; ++L) {
        for (int l = 0; l < 3; ++l) {
            const float* wnl = (const float*)d_in[5 + 6 * L + l];
            const float* wrl = (const float*)d_in[5 + 6 * L + 3 + l];
            k_prep_w<<<(CC * tau_nl[l] + 255) / 256, 256, 0, stream>>>(
                wnl, WT16 + L * WT_LAYER + off_nl[l], tau_nl[l], tau_nl[l]);
            k_prep_w<<<(CC * taup_rl[l] + 255) / 256, 256, 0, stream>>>(
                wrl, WT16 + L * WT_LAYER + off_rl[l], tau_rl[l], taup_rl[l]);
        }
    }

    k_pack<<<(NODES * FDIM + 255) / 256, 256, 0, stream>>>(vert0, vert1, vert2, P, PT16);
    k_ysum<<<NODES, 64, 0, stream>>>(rel, ysum);

    for (int L = 0; L < 2; ++L) {
        k_msgpass<<<BB * 12 * 9, 32, 0, stream>>>(conn16, PT16, mp);
        k_node<<<NODES, 96, 0, stream>>>(mp, ysum, cg, WT16 + L * WT_LAYER, ps);
        k_norm<<<3, 256, 0, stream>>>(ps, scale);
        k_scale<<<(NODES * FDIM + 255) / 256, 256, 0, stream>>>(ps, scale, P, PT16);
        k_scalars<<<NODES, 256, 0, stream>>>(P, out, L);
    }
}
